// MMGLT_22978075034362
// MI455X (gfx1250) — matrix operand order repaired and checked by reading, not compiled, not run on hardware
//
#include <hip/hip_runtime.h>

// ---------------------------------------------------------------------------
// Types / helpers
// ---------------------------------------------------------------------------
typedef __bf16 bf16_t;
typedef __attribute__((ext_vector_type(16))) __bf16   v16bf;
typedef __attribute__((ext_vector_type(8)))  float    v8f;
typedef __attribute__((ext_vector_type(8)))  unsigned v8u;
typedef int v4i __attribute__((__vector_size__(16)));   // matches async builtin pointee

#define ASG __attribute__((address_space(1)))
#define ASL __attribute__((address_space(3)))

#define DEV static __device__ __forceinline__

DEV unsigned short bf_bits(bf16_t h) { return __builtin_bit_cast(unsigned short, h); }
DEV bf16_t f2bf(float f) {
  unsigned u = __builtin_bit_cast(unsigned, f);
  u += 0x7FFFu + ((u >> 16) & 1u);                  // round-to-nearest-even
  return __builtin_bit_cast(bf16_t, (unsigned short)(u >> 16));
}
DEV unsigned pack_bf2(float lo, float hi) {
  return (unsigned)bf_bits(f2bf(lo)) | ((unsigned)bf_bits(f2bf(hi)) << 16);
}
DEV void store_out(float*  p, float v) { *p = v; }
DEV void store_out(bf16_t* p, float v) { *p = f2bf(v); }

DEV v8f wmma_bf16(v16bf a, v16bf b, v8f c) {
  // v_wmma_f32_16x16x32_bf16  (neg_a, A, neg_b, B, c_mod, C, reuse_a, reuse_b)
  return __builtin_amdgcn_wmma_f32_16x16x32_bf16(false, a, false, b, (short)0, c,
                                                 false, false);
}

DEV v16bf pack_frag(uint4 lo, uint4 hi) {
  v8u t;
  t[0] = lo.x; t[1] = lo.y; t[2] = lo.z; t[3] = lo.w;
  t[4] = hi.x; t[5] = hi.y; t[6] = hi.z; t[7] = hi.w;
  return __builtin_bit_cast(v16bf, t);
}

// A-fragment, direct from a row-major bf16 row pointer (global or LDS).
// p must already include: row*ld + ((lane<16)?0:8) + k0.
// ISA layout: per lane two contiguous K runs: [base..base+7] and [base+16..23].
DEV v16bf load_a_direct(const bf16_t* p) {
  return pack_frag(*(const uint4*)p, *(const uint4*)(p + 16));
}

// B-fragment: per lane two contiguous K runs: [base..base+7] and [base+16..23].
// p must already include: col*ld + ((lane<16)?0:8) + k0.
DEV v16bf load_b_frag(const bf16_t* p) {
  const uint4* q = (const uint4*)p;
  return pack_frag(q[0], q[2]);
}

DEV void wait_async0() {
#if __has_builtin(__builtin_amdgcn_s_wait_asynccnt)
  __builtin_amdgcn_s_wait_asynccnt(0);
#else
  asm volatile("s_wait_asynccnt 0x0" ::: "memory");
#endif
}

// ---------------------------------------------------------------------------
// Kernel 1: weight repack f32 -> bf16 (conv weight transposed to patch-K order)
//   srwb[o*4096 + (kh*4+kw)*256 + i] = sr_w[o][i][kh][kw]
// ---------------------------------------------------------------------------
__global__ __launch_bounds__(256) void pack_weights_kernel(
    const float* __restrict__ qw, const float* __restrict__ kvw,
    const float* __restrict__ pw, const float* __restrict__ srw,
    bf16_t* __restrict__ qwb, bf16_t* __restrict__ kvwb,
    bf16_t* __restrict__ pwb, bf16_t* __restrict__ srwb)
{
  const int idx = blockIdx.x * 256 + threadIdx.x;
  if (idx < 65536)   qwb[idx]  = f2bf(qw[idx]);
  if (idx < 131072)  kvwb[idx] = f2bf(kvw[idx]);
  if (idx < 65536)   pwb[idx]  = f2bf(pw[idx]);
  if (idx < 1048576) {
    const int o   = idx >> 12;
    const int rem = idx & 4095;
    const int s   = rem >> 8;     // kh*4+kw
    const int i   = rem & 255;    // input channel
    srwb[idx] = f2bf(srw[(o << 12) + (i << 4) + s]);
  }
}

// ---------------------------------------------------------------------------
// Kernel 2: one-shot activation conversion f32 -> bf16 (paired u32 stores).
//   Done ONCE per input so every GEMM consumes bf16 directly (no re-convert).
// ---------------------------------------------------------------------------
__global__ __launch_bounds__(256) void cvt_f32_bf16_kernel(
    const float* __restrict__ src, bf16_t* __restrict__ dst, int npairs)
{
  const int idx = blockIdx.x * 256 + threadIdx.x;
  if (idx < npairs) {
    const float2 v = *(const float2*)(src + 2 * (size_t)idx);
    ((unsigned*)dst)[idx] = pack_bf2(v.x, v.y);
  }
}

// ---------------------------------------------------------------------------
// Kernel 3: WMMA GEMM, bf16 A — fragments loaded DIRECTLY from global
//   (two b128 loads per A fragment; no LDS, no barriers, no conversion VALU)
//   Out[M,ldOut] (+bias) = A[M,K] * W^T ;  W row-major [ldOut x K] bf16
// ---------------------------------------------------------------------------
template <typename TOUT>
__global__ __launch_bounds__(128) void wmma_gemm_bf16a_kernel(
    const bf16_t* __restrict__ A, const bf16_t* __restrict__ W,
    TOUT* __restrict__ Out, const float* __restrict__ bias,
    int Mrows, int K, int ldOut)
{
  const int tid  = threadIdx.x;
  const int wave = tid >> 5;
  const int lane = tid & 31;
  const int mBase = blockIdx.x * 64;
  const int nBase = blockIdx.y * 32;

  // A row this lane feeds; clamp out-of-range rows (their C rows are unstored)
  int arow = mBase + wave * 16 + (lane & 15);
  if (arow >= Mrows) arow = Mrows - 1;
  const bf16_t* aPtr = A + (size_t)arow * K + ((lane < 16) ? 0 : 8);
  const bf16_t* w0   = W + (size_t)(nBase + (lane & 15)) * K + ((lane < 16) ? 0 : 8);
  const bf16_t* w1   = w0 + (size_t)16 * K;

  v8f acc0 = {0.f, 0.f, 0.f, 0.f, 0.f, 0.f, 0.f, 0.f};
  v8f acc1 = acc0;

  for (int k0 = 0; k0 < K; k0 += 32) {
    const v16bf af = load_a_direct(aPtr + k0);
    const v16bf b0 = load_b_frag(w0 + k0);
    const v16bf b1 = load_b_frag(w1 + k0);
    acc0 = wmma_bf16(af, b0, acc0);
    acc1 = wmma_bf16(af, b1, acc1);
  }

  const int   rBase = mBase + wave * 16 + ((lane < 16) ? 0 : 8);
  const int   col   = nBase + (lane & 15);
  const float bi0   = bias ? bias[col] : 0.f;
  const float bi1   = bias ? bias[col + 16] : 0.f;
#pragma unroll
  for (int rr = 0; rr < 8; ++rr) {
    const int row = rBase + rr;
    if (row < Mrows) {
      store_out(Out + (size_t)row * ldOut + col,      acc0[rr] + bi0);
      store_out(Out + (size_t)row * ldOut + col + 16, acc1[rr] + bi1);
    }
  }
}

// ---------------------------------------------------------------------------
// Kernel 4: SR conv as DIRECT patch GEMM on the bf16 image
//   Out[784,256] f32 = patches[784,4096] * srwb^T + bias
//   patch K order: k = (kh*4+kw)*256 + i  -> every 32-wide K chunk is a
//   contiguous channel run at one spatial position => direct b128 A fragments.
// ---------------------------------------------------------------------------
__global__ __launch_bounds__(128) void srconv_kernel(
    const bf16_t* __restrict__ Xb, const bf16_t* __restrict__ Wsr,
    const float* __restrict__ bias, float* __restrict__ Out)
{
  const int tid  = threadIdx.x;
  const int wave = tid >> 5;
  const int lane = tid & 31;
  const int mBase = blockIdx.x * 64;
  const int nBase = blockIdx.y * 32;

  int arow = mBase + wave * 16 + (lane & 15);
  if (arow >= 784) arow = 783;                       // clamp (rows unstored)
  const int bb = arow / 196;
  const int p  = arow % 196;
  const int ph = p / 14, pw = p % 14;
  const bf16_t* aBase = Xb +
      (size_t)(bb * 3136 + (4 * ph) * 56 + 4 * pw) * 256 + ((lane < 16) ? 0 : 8);
  const bf16_t* w0 = Wsr + (size_t)(nBase + (lane & 15)) * 4096 + ((lane < 16) ? 0 : 8);
  const bf16_t* w1 = w0 + (size_t)16 * 4096;

  v8f acc0 = {0.f, 0.f, 0.f, 0.f, 0.f, 0.f, 0.f, 0.f};
  v8f acc1 = acc0;

  for (int kc = 0; kc < 128; ++kc) {                 // K = 4096
    const int k0 = kc * 32;
    const int s  = k0 >> 8;                          // kh*4+kw
    const int i0 = k0 & 255;                         // channel base
    const int kh = s >> 2, kw = s & 3;
    const v16bf af = load_a_direct(aBase + (size_t)(kh * 56 + kw) * 256 + i0);
    const v16bf b0 = load_b_frag(w0 + k0);
    const v16bf b1 = load_b_frag(w1 + k0);
    acc0 = wmma_bf16(af, b0, acc0);
    acc1 = wmma_bf16(af, b1, acc1);
  }

  const int   rBase = mBase + wave * 16 + ((lane < 16) ? 0 : 8);
  const int   col   = nBase + (lane & 15);
  const float bi0   = bias[col];
  const float bi1   = bias[col + 16];
#pragma unroll
  for (int rr = 0; rr < 8; ++rr) {
    const int row = rBase + rr;
    if (row < 784) {
      Out[(size_t)row * 256 + col]      = acc0[rr] + bi0;
      Out[(size_t)row * 256 + col + 16] = acc1[rr] + bi1;
    }
  }
}

// ---------------------------------------------------------------------------
// Kernel 5: LayerNorm over C=256, one wave32 per row, bf16 output
// ---------------------------------------------------------------------------
__global__ __launch_bounds__(256) void ln_kernel(
    const float* __restrict__ T, const float* __restrict__ G,
    const float* __restrict__ Bv, bf16_t* __restrict__ Out)
{
  const int row  = blockIdx.x * 8 + (threadIdx.x >> 5);
  const int lane = threadIdx.x & 31;
  const float* src = T + (size_t)row * 256;
  float v[8];
  float sum = 0.f;
#pragma unroll
  for (int j = 0; j < 8; ++j) { v[j] = src[j * 32 + lane]; sum += v[j]; }
#pragma unroll
  for (int m = 1; m < 32; m <<= 1) sum += __shfl_xor(sum, m, 32);
  const float mu = sum * (1.f / 256.f);
  float var = 0.f;
#pragma unroll
  for (int j = 0; j < 8; ++j) { const float d = v[j] - mu; var += d * d; }
#pragma unroll
  for (int m = 1; m < 32; m <<= 1) var += __shfl_xor(var, m, 32);
  const float rstd = rsqrtf(var * (1.f / 256.f) + 1e-5f);
  bf16_t* drow = Out + (size_t)row * 256;
#pragma unroll
  for (int j = 0; j < 8; ++j) {
    const int c = j * 32 + lane;
    drow[c] = f2bf((v[j] - mu) * rstd * G[c] + Bv[c]);
  }
}

// ---------------------------------------------------------------------------
// Kernel 6: fused attention for one (b,h): softmax(Q K^T * ls) V
//   Q: bf16 [B*3136, 256] (head slice h*32..); KV: bf16 [B*196, 512]
//   Block 128 thr = 4 waves, 64 q-rows/block; kv len 196 padded to 208/224.
//   K tile staged with GLOBAL_LOAD_ASYNC_TO_LDS_B128 when available.
// ---------------------------------------------------------------------------
__global__ __launch_bounds__(128) void attn_kernel(
    const bf16_t* __restrict__ Q, const bf16_t* __restrict__ KB,
    const bf16_t* __restrict__ VB, bf16_t* __restrict__ O, float lscale)
{
  __shared__ bf16_t k_lds[208 * 32];       // [kv_row][kd] row-major
  __shared__ bf16_t vT_lds[32 * 224];      // [d][kv_row] (transposed V)
  __shared__ bf16_t p_lds[4][16 * 224];    // per-wave probabilities

  const int tid  = threadIdx.x;
  const int wave = tid >> 5;
  const int lane = tid & 31;
  const int b = (int)blockIdx.y >> 3;
  const int h = (int)blockIdx.y & 7;
  const int qTile = blockIdx.x * 64 + wave * 16;
  const bf16_t z0 = f2bf(0.f);

  // ---- stage K tile: async DMA to LDS (b128 = 8 bf16 per lane-op) ----
#if __has_builtin(__builtin_amdgcn_global_load_async_to_lds_b128)
  for (int idx = tid; idx < 208 * 4; idx += 128) {   // 16B chunks
    const int n = idx >> 2;                          // kv row 0..207
    const int c = (idx & 3) * 8;                     // element offset in row
    bf16_t* dst = k_lds + n * 32 + c;
    if (n < 196) {
      const bf16_t* src = KB + (size_t)(b * 196 + n) * 512 + h * 32 + c;
      __builtin_amdgcn_global_load_async_to_lds_b128(
          (ASG v4i*)(unsigned long long)src,
          (ASL v4i*)(unsigned)(unsigned long long)dst, 0, 0);
    } else {
      const uint4 zz = {0u, 0u, 0u, 0u};
      *(uint4*)dst = zz;                             // zero padding rows
    }
  }
#else
  for (int idx = tid; idx < 208 * 32; idx += 128) {
    const int n = idx >> 5, kd = idx & 31;
    k_lds[idx] = (n < 196) ? KB[(size_t)(b * 196 + n) * 512 + h * 32 + kd] : z0;
  }
#endif
  // ---- stage V^T (transpose: per-element path) ----
  for (int idx = tid; idx < 32 * 224; idx += 128) {
    const int d = idx / 224, kr = idx % 224;
    vT_lds[idx] = (kr < 196) ? VB[(size_t)(b * 196 + kr) * 512 + 256 + h * 32 + d] : z0;
  }
#if __has_builtin(__builtin_amdgcn_global_load_async_to_lds_b128)
  wait_async0();
#endif
  __syncthreads();

  // Q A-fragment (HD = 32 = exactly one K chunk), two b128 loads from global
  const v16bf qf = load_a_direct(
      Q + (size_t)(b * 3136 + qTile + (lane & 15)) * 256 + h * 32 + ((lane < 16) ? 0 : 8));

  // logits: 16 q-rows x 208 kv-cols = 13 WMMA accumulators
  v8f s[13];
  const int kbH = (lane < 16) ? 0 : 8;
#pragma unroll
  for (int c = 0; c < 13; ++c) {
    v8f zacc = {0.f, 0.f, 0.f, 0.f, 0.f, 0.f, 0.f, 0.f};
    const v16bf kf = load_b_frag(k_lds + (size_t)(c * 16 + (lane & 15)) * 32 + kbH);
    s[c] = wmma_bf16(qf, kf, zacc);
  }

  // scale + mask padded cols (196..207 live in chunk 12, (lane&15)>=4)
  const bool pad = (lane & 15) >= 4;
#pragma unroll
  for (int c = 0; c < 13; ++c)
#pragma unroll
    for (int i = 0; i < 8; ++i) {
      float v = s[c][i] * lscale;
      if (c == 12) v = pad ? -1e30f : v;
      s[c][i] = v;
    }

  // row softmax; row M = r + 8*(lane>=16), cols spread over 16 lanes of a half
  // -> xor masks 1,2,4,8 reduce exactly one row (stay inside the half)
#pragma unroll
  for (int r = 0; r < 8; ++r) {
    float m = -1e30f;
#pragma unroll
    for (int c = 0; c < 13; ++c) m = fmaxf(m, s[c][r]);
#pragma unroll
    for (int msk = 1; msk < 16; msk <<= 1) m = fmaxf(m, __shfl_xor(m, msk, 32));
    float sum = 0.f;
#pragma unroll
    for (int c = 0; c < 13; ++c) {
      const float e = __expf(s[c][r] - m);
      s[c][r] = e;
      sum += e;
    }
#pragma unroll
    for (int msk = 1; msk < 16; msk <<= 1) sum += __shfl_xor(sum, msk, 32);
    const float inv = 1.f / sum;
#pragma unroll
    for (int c = 0; c < 13; ++c) s[c][r] *= inv;
  }

  // C-layout -> A-layout via per-wave LDS tile (cols 208..223 zero pad)
  {
    bf16_t* pwt = p_lds[wave];
    const int colL = lane & 15;
    const int rOff = (lane < 16) ? 0 : 8;
#pragma unroll
    for (int r = 0; r < 8; ++r) {
      const int rowp = (rOff + r) * 224;
#pragma unroll
      for (int c = 0; c < 13; ++c) pwt[rowp + c * 16 + colL] = f2bf(s[c][r]);
      pwt[rowp + 208 + colL] = z0;
    }
  }
  __syncthreads();

  // O = P x V   (K = 224 in 7 chunks of 32; padded P rows are zero)
  v8f o0 = {0.f, 0.f, 0.f, 0.f, 0.f, 0.f, 0.f, 0.f};
  v8f o1 = o0;
  const bf16_t* pwt = p_lds[wave];
  const bf16_t* pA  = pwt + (size_t)(lane & 15) * 224 + ((lane < 16) ? 0 : 8);
#pragma unroll
  for (int kc = 0; kc < 7; ++kc) {
    const v16bf af = load_a_direct(pA + kc * 32);          // 2x ds_load_b128
    const int   kb = kc * 32 + kbH;
    const v16bf b0 = load_b_frag(vT_lds + (size_t)(lane & 15) * 224 + kb);
    const v16bf b1 = load_b_frag(vT_lds + (size_t)(16 + (lane & 15)) * 224 + kb);
    o0 = wmma_bf16(af, b0, o0);
    o1 = wmma_bf16(af, b1, o1);
  }

  const int rOff = (lane < 16) ? 0 : 8;
  const int d0   = lane & 15;
#pragma unroll
  for (int r = 0; r < 8; ++r) {
    const size_t row = (size_t)(b * 3136 + qTile + rOff + r);
    O[row * 256 + h * 32 + d0]      = f2bf(o0[r]);
    O[row * 256 + h * 32 + 16 + d0] = f2bf(o1[r]);
  }
}

// ---------------------------------------------------------------------------
// Host orchestration
// ---------------------------------------------------------------------------
extern "C" void kernel_launch(void* const* d_in, const int* in_sizes, int n_in,
                              void* d_out, int out_size, void* d_ws, size_t ws_size,
                              hipStream_t stream) {
  (void)in_sizes; (void)n_in; (void)out_size; (void)ws_size;
  const float* x      = (const float*)d_in[0];
  const float* y      = (const float*)d_in[1];
  const float* z      = (const float*)d_in[2];
  const float* q_w    = (const float*)d_in[3];
  const float* kv_w   = (const float*)d_in[4];
  const float* sr_w   = (const float*)d_in[5];
  const float* sr_b   = (const float*)d_in[6];
  const float* ln_g   = (const float*)d_in[7];
  const float* ln_b   = (const float*)d_in[8];
  const float* proj_w = (const float*)d_in[9];
  const float* proj_b = (const float*)d_in[10];
  float* out = (float*)d_out;

  char*  ws  = (char*)d_ws;
  size_t off = 0;
  auto take = [&](size_t bytes) -> void* {
    void* p = ws + off;
    off = (off + bytes + 255) & ~(size_t)255;
    return p;
  };

  bf16_t* qwb    = (bf16_t*)take((size_t)65536 * 2);
  bf16_t* kvwb   = (bf16_t*)take((size_t)131072 * 2);
  bf16_t* projwb = (bf16_t*)take((size_t)65536 * 2);
  bf16_t* srwb   = (bf16_t*)take((size_t)1048576 * 2);
  bf16_t* ab[3];  for (int i = 0; i < 3; ++i) ab[i]  = (bf16_t*)take((size_t)3211264 * 2); // x,y,z bf16
  bf16_t* qb[3];  for (int i = 0; i < 3; ++i) qb[i]  = (bf16_t*)take((size_t)3211264 * 2);
  float*  tb[3];  for (int i = 0; i < 3; ++i) tb[i]  = (float*) take((size_t)200704 * 4);
  bf16_t* lnbuf[3]; for (int i = 0; i < 3; ++i) lnbuf[i] = (bf16_t*)take((size_t)200704 * 2);
  bf16_t* kvb[3]; for (int i = 0; i < 3; ++i) kvb[i] = (bf16_t*)take((size_t)401408 * 2);
  bf16_t* ob[3];  for (int i = 0; i < 3; ++i) ob[i]  = (bf16_t*)take((size_t)3211264 * 2);

  // 1) weights -> bf16 (+ conv weight transposed to patch-K order)
  pack_weights_kernel<<<4096, 256, 0, stream>>>(q_w, kv_w, proj_w, sr_w,
                                                qwb, kvwb, projwb, srwb);

  // 2) activations -> bf16, once (all downstream GEMMs consume bf16 directly)
  cvt_f32_bf16_kernel<<<6272, 256, 0, stream>>>(x, ab[0], 1605632);
  cvt_f32_bf16_kernel<<<6272, 256, 0, stream>>>(y, ab[1], 1605632);
  cvt_f32_bf16_kernel<<<6272, 256, 0, stream>>>(z, ab[2], 1605632);

  // 3) Q projections (scales folded into attention logit scale)
  const dim3 gQ(196, 8);
  wmma_gemm_bf16a_kernel<bf16_t><<<gQ, 128, 0, stream>>>(ab[0], qwb, qb[0], nullptr, 12544, 256, 256);
  wmma_gemm_bf16a_kernel<bf16_t><<<gQ, 128, 0, stream>>>(ab[1], qwb, qb[1], nullptr, 12544, 256, 256);
  wmma_gemm_bf16a_kernel<bf16_t><<<gQ, 128, 0, stream>>>(ab[2], qwb, qb[2], nullptr, 12544, 256, 256);

  // 4) SR conv as direct patch GEMM: tb[0]<-x, tb[1]<-z, tb[2]<-y
  const dim3 gC(13, 8);
  srconv_kernel<<<gC, 128, 0, stream>>>(ab[0], srwb, sr_b, tb[0]);
  srconv_kernel<<<gC, 128, 0, stream>>>(ab[2], srwb, sr_b, tb[1]);
  srconv_kernel<<<gC, 128, 0, stream>>>(ab[1], srwb, sr_b, tb[2]);

  // 5) LayerNorm -> bf16
  ln_kernel<<<98, 256, 0, stream>>>(tb[0], ln_g, ln_b, lnbuf[0]);
  ln_kernel<<<98, 256, 0, stream>>>(tb[1], ln_g, ln_b, lnbuf[1]);
  ln_kernel<<<98, 256, 0, stream>>>(tb[2], ln_g, ln_b, lnbuf[2]);

  // 6) KV projection [784,256] x [256,512]  (direct bf16-A WMMA GEMM)
  const dim3 gKV(13, 16);
  wmma_gemm_bf16a_kernel<bf16_t><<<gKV, 128, 0, stream>>>(lnbuf[0], kvwb, kvb[0], nullptr, 784, 256, 512);
  wmma_gemm_bf16a_kernel<bf16_t><<<gKV, 128, 0, stream>>>(lnbuf[1], kvwb, kvb[1], nullptr, 784, 256, 512);
  wmma_gemm_bf16a_kernel<bf16_t><<<gKV, 128, 0, stream>>>(lnbuf[2], kvwb, kvb[2], nullptr, 784, 256, 512);

  // 7) attention (qx*0.5 and kx*0.5 folded into logit scale)
  const float SC = 0.17677669529663687f;   // 32^-0.5
  const dim3 gA(49, 32);
  attn_kernel<<<gA, 128, 0, stream>>>(qb[0], kvb[1], kvb[1], ob[0], 0.5f * SC); // x_out
  attn_kernel<<<gA, 128, 0, stream>>>(qb[1], kvb[1], kvb[1], ob[1], SC);        // y_out
  attn_kernel<<<gA, 128, 0, stream>>>(qb[2], kvb[0], kvb[2], ob[2], 0.5f * SC); // z_out

  // 8) output projections -> d_out (tuple order x,y,z)
  wmma_gemm_bf16a_kernel<float><<<gQ, 128, 0, stream>>>(ob[0], projwb, out,               proj_b, 12544, 256, 256);
  wmma_gemm_bf16a_kernel<float><<<gQ, 128, 0, stream>>>(ob[1], projwb, out + 3211264,     proj_b, 12544, 256, 256);
  wmma_gemm_bf16a_kernel<float><<<gQ, 128, 0, stream>>>(ob[2], projwb, out + 2 * 3211264, proj_b, 12544, 256, 256);
}
